// LSTM_Model_37074157699391
// MI455X (gfx1250) — matrix operand order repaired and checked by reading, not compiled, not run on hardware
//
#include <hip/hip_runtime.h>
#include <math.h>

typedef __attribute__((ext_vector_type(16))) _Float16 v16h;
typedef __attribute__((ext_vector_type(8)))  float    v8f;

#define HDIM 51
#define G4   204     // 4*H gate rows
#define GPAD 256     // 16 tiles * 16 rows (waves 0..7 own tiles {w, w+8})

__device__ __forceinline__ float sigf(float v) { return 1.0f / (1.0f + __expf(-v)); }

#define WMMA_F16(A, B, C) \
  __builtin_amdgcn_wmma_f32_16x16x32_f16(false, (A), false, (B), (short)0, (C), false, false)

__global__ __launch_bounds__(256, 1)
void lstm2_persistent(const float* __restrict__ x,
                      const float* __restrict__ Wih1,
                      const float* __restrict__ Whh1,
                      const float* __restrict__ bih1,
                      const float* __restrict__ bhh1,
                      const float* __restrict__ Wih2,
                      const float* __restrict__ Whh2,
                      const float* __restrict__ bih2,
                      const float* __restrict__ bhh2,
                      const float* __restrict__ Wlin,
                      const float* __restrict__ blin,
                      float* __restrict__ out,
                      int T)
{
  // hcat: [0,64) = h1 (fp16, zero-padded 51->64), [64,128) = h2, [128,160) = permanent zeros
  // (zero region feeds the B-fragment lanes whose matrix column must be 0)
  __shared__ __align__(32) _Float16 hcat[160];
  __shared__ float gates1[GPAD];
  __shared__ float gates2[GPAD];
  __shared__ float b1s[GPAD], b2s[GPAD], wih1s[GPAD];
  __shared__ float wlins[64];
  __shared__ float pbuf[64];
  __shared__ float blins;

  const int tid  = (int)threadIdx.x;
  const int wave = tid >> 5;
  const int lane = tid & 31;
  const int hi   = lane >> 4;   // 0: lanes 0-15, 1: lanes 16-31
  const int llo  = lane & 15;

  // ---------------- prologue (runs once; not on the recurrence critical path) ---------------
  b1s[tid]   = (tid < G4) ? (bih1[tid] + bhh1[tid]) : 0.f;
  b2s[tid]   = (tid < G4) ? (bih2[tid] + bhh2[tid]) : 0.f;
  wih1s[tid] = (tid < G4) ? Wih1[tid] : 0.f;
  if (tid < 64) { wlins[tid] = (tid < HDIM) ? Wlin[tid] : 0.f; pbuf[tid] = 0.f; }
  if (tid < 160) hcat[tid] = (_Float16)0.f;
  if (tid == 0) blins = blin[0];

  // Pack weight rows into WMMA A-fragments (fp16 16x32, contiguous k order, same as B:
  // lane holds row M = tile*16 + (lane&15); element e -> K = kt*32 + hi*16 + e).
  // These live in VGPRs for the entire 262144-step recurrence.
  const int m0 = wave, m1 = wave + 8;  // tiles {w, w+8} cover rows 0..255 exactly once
  v16h a1[2][2];   // layer-1: Whh1 (204x51 -> 208x64), 2 K-tiles
  v16h a2[2][4];   // layer-2: [Wih2 | Whh2] (204x102 -> 208x128), 4 K-tiles
  #pragma unroll
  for (int ti = 0; ti < 2; ++ti) {
    const int  row = (ti ? m1 : m0) * 16 + llo;
    const bool rok = row < G4;
    #pragma unroll
    for (int kt = 0; kt < 2; ++kt) {
      v16h v;
      #pragma unroll
      for (int e = 0; e < 16; ++e) {
        const int k = kt * 32 + hi * 16 + e;
        const float w = (rok && k < HDIM) ? Whh1[row * HDIM + k] : 0.f;
        v[e] = (_Float16)w;
      }
      a1[ti][kt] = v;
    }
    #pragma unroll
    for (int kt = 0; kt < 4; ++kt) {
      v16h v;
      #pragma unroll
      for (int e = 0; e < 16; ++e) {
        const int k = kt * 32 + hi * 16 + e;
        float w = 0.f;
        if (rok) {
          if (k < HDIM)                        w = Wih2[row * HDIM + k];
          else if (k >= 64 && k < 64 + HDIM)   w = Whh2[row * HDIM + (k - 64)];
        }
        v[e] = (_Float16)w;
      }
      a2[ti][kt] = v;
    }
  }

  float c1 = 0.f, c2 = 0.f;       // cell state lives in registers of threads 0..50
  __syncthreads();

  const int r0 = m0 * 16 + hi * 8;  // gate rows written by lanes 0/16 (D col N=0 holders)
  const int r1 = m1 * 16 + hi * 8;

  // ---------------- persistent recurrence ----------------
  for (int t = 0; t < T; ++t) {
    // Phase A: layer-1 gates = Whh1 * h1  (vector in B column 0)
    {
      v8f acc0 = {}; v8f acc1 = {};
      #pragma unroll
      for (int kt = 0; kt < 2; ++kt) {
        const int  boff = (llo == 0) ? (kt * 32 + hi * 16) : 128;  // 128 => zero pad
        const v16h b    = *(const v16h*)&hcat[boff];
        acc0 = WMMA_F16(a1[0][kt], b, acc0);
        acc1 = WMMA_F16(a1[1][kt], b, acc1);
      }
      if (llo == 0) {                 // lanes 0/16 hold D column 0 (8 rows each)
        const float xt = x[t];
        #pragma unroll
        for (int i = 0; i < 8; ++i)
          gates1[r0 + i] = acc0[i] + xt * wih1s[r0 + i] + b1s[r0 + i];
        #pragma unroll
        for (int i = 0; i < 8; ++i)
          gates1[r1 + i] = acc1[i] + xt * wih1s[r1 + i] + b1s[r1 + i];
      }
    }
    __syncthreads();

    // Cell update 1 (gate order i,f,g,o)
    if (tid < HDIM) {
      const float gi = gates1[tid],            gf = gates1[HDIM + tid];
      const float gg = gates1[2 * HDIM + tid], go = gates1[3 * HDIM + tid];
      c1 = sigf(gf) * c1 + sigf(gi) * tanhf(gg);
      const float h1v = sigf(go) * tanhf(c1);
      hcat[tid] = (_Float16)h1v;
    }
    __syncthreads();

    // Phase B: layer-2 gates = [Wih2|Whh2] * [h1;h2]
    {
      v8f acc0 = {}; v8f acc1 = {};
      #pragma unroll
      for (int kt = 0; kt < 4; ++kt) {
        const int  boff = (llo == 0) ? (kt * 32 + hi * 16) : 128;
        const v16h b    = *(const v16h*)&hcat[boff];
        acc0 = WMMA_F16(a2[0][kt], b, acc0);
        acc1 = WMMA_F16(a2[1][kt], b, acc1);
      }
      if (llo == 0) {
        #pragma unroll
        for (int i = 0; i < 8; ++i) gates2[r0 + i] = acc0[i] + b2s[r0 + i];
        #pragma unroll
        for (int i = 0; i < 8; ++i) gates2[r1 + i] = acc1[i] + b2s[r1 + i];
      }
    }
    __syncthreads();

    // Cell update 2 + per-element output partials
    if (tid < HDIM) {
      const float gi = gates2[tid],            gf = gates2[HDIM + tid];
      const float gg = gates2[2 * HDIM + tid], go = gates2[3 * HDIM + tid];
      c2 = sigf(gf) * c2 + sigf(gi) * tanhf(gg);
      const float h2v = sigf(go) * tanhf(c2);
      hcat[64 + tid] = (_Float16)h2v;
      pbuf[tid] = h2v * wlins[tid];
    }
    __syncthreads();

    // out[t] = Wlin . h2 + blin  (wave-0 shuffle reduction; pbuf[51..63] stay zero)
    if (wave == 0) {
      float s = pbuf[lane] + pbuf[lane + 32];
      s += __shfl_xor(s, 16, 32);
      s += __shfl_xor(s,  8, 32);
      s += __shfl_xor(s,  4, 32);
      s += __shfl_xor(s,  2, 32);
      s += __shfl_xor(s,  1, 32);
      if (lane == 0) out[t] = s + blins;
    }
  }
}

extern "C" void kernel_launch(void* const* d_in, const int* in_sizes, int n_in,
                              void* d_out, int out_size, void* d_ws, size_t ws_size,
                              hipStream_t stream) {
  const float* xp    = (const float*)d_in[0];
  const float* Wih1  = (const float*)d_in[1];
  const float* Whh1  = (const float*)d_in[2];
  const float* bih1  = (const float*)d_in[3];
  const float* bhh1  = (const float*)d_in[4];
  const float* Wih2  = (const float*)d_in[5];
  const float* Whh2  = (const float*)d_in[6];
  const float* bih2  = (const float*)d_in[7];
  const float* bhh2  = (const float*)d_in[8];
  const float* Wlin  = (const float*)d_in[9];
  const float* blinp = (const float*)d_in[10];
  float* out = (float*)d_out;
  const int T = in_sizes[0];  // 262144

  lstm2_persistent<<<1, 256, 0, stream>>>(xp, Wih1, Whh1, bih1, bhh1,
                                          Wih2, Whh2, bih2, bhh2,
                                          Wlin, blinp, out, T);
}
